// ContrastiveLoss_71279277244723
// MI455X (gfx1250) — matrix operand order repaired and checked by reading, not compiled, not run on hardware
//
#include <hip/hip_runtime.h>
#include <hip/hip_bf16.h>
#include <math.h>

typedef __attribute__((ext_vector_type(16))) __bf16 v16bf;
typedef __attribute__((ext_vector_type(8)))  float  v8f;

#define BSZ 8192
#define DDIM 128
#define NCHUNK 16
#define COLS_PER_CHUNK (BSZ / NCHUNK)   // 512
#define NTILES (COLS_PER_CHUNK / 16)    // 32
#define LDSB_STRIDE 136                 // bf16 elems: 68 dwords/row -> minimal bank conflicts
#define INV_TEMP 10.0f
#define NEG_BIG (-1e30f)

union FragBF {
    v16bf  v;
    float4 f[2];
};

// ---------------- kernel 0: zero the scalar output ----------------
__global__ void zero_out_kernel(float* out) {
    if (threadIdx.x == 0) out[0] = 0.0f;
}

// ---------------- kernel 1: split f32 -> bf16 hi + bf16 lo (RNE, bit math) ----------------
__global__ __launch_bounds__(256) void split_bf16_kernel(const float* __restrict__ X,
                                                         unsigned short* __restrict__ hi,
                                                         unsigned short* __restrict__ lo) {
    const size_t i0 = ((size_t)blockIdx.x * 256 + threadIdx.x) * 4;
    const float4 x = *(const float4*)(X + i0);
    float xs[4] = {x.x, x.y, x.z, x.w};
#pragma unroll
    for (int j = 0; j < 4; ++j) {
        const unsigned int u  = __float_as_uint(xs[j]);
        const unsigned int rh = (u + 0x7FFFu + ((u >> 16) & 1u)) & 0xFFFF0000u;
        const float        hf = __uint_as_float(rh);
        const float        lf = xs[j] - hf;
        const unsigned int ul = __float_as_uint(lf);
        const unsigned int rl = (ul + 0x7FFFu + ((ul >> 16) & 1u)) & 0xFFFF0000u;
        hi[i0 + j] = (unsigned short)(rh >> 16);
        lo[i0 + j] = (unsigned short)(rl >> 16);
    }
}

// ---------------- kernel 2: per-row squared L2 norms (from exact f32) ----------------
__global__ __launch_bounds__(256) void row_norms_kernel(const float* __restrict__ X,
                                                        float* __restrict__ out) {
    const int wave = threadIdx.x >> 5;
    const int lane = threadIdx.x & 31;
    const int row  = blockIdx.x * 8 + wave;
    const float4 x = ((const float4*)(X + (size_t)row * DDIM))[lane];
    float acc = x.x * x.x + x.y * x.y + x.z * x.z + x.w * x.w;
#pragma unroll
    for (int m = 16; m >= 1; m >>= 1) acc += __shfl_xor(acc, m);
    if (lane == 0) out[row] = acc;
}

// online-softmax merge of one new logit l into (m,s): single exp
__device__ __forceinline__ void online_update(float& m, float& s, float l) {
    const float diff = l - m;
    const float e    = __expf(-fabsf(diff));      // exp(-(nm - other))
    s = (diff > 0.0f) ? __fmaf_rn(s, e, 1.0f) : (s + e);
    m = fmaxf(m, l);
}

// merge two (m,s) states: single exp
__device__ __forceinline__ void online_merge(float& m, float& s, float om, float os) {
    const float diff = om - m;
    const float e    = __expf(-fabsf(diff));
    s = (diff > 0.0f) ? __fmaf_rn(s, e, os) : __fmaf_rn(os, e, s);
    m = fmaxf(m, om);
}

// ---------------- kernel 3: fused bf16x3 WMMA GEMM + online softmax ----------------
// grid: (64 row-blocks of 128 rows, 16 column chunks of 512 cols); block 256 = 8 waves
__global__ __launch_bounds__(256) void gemm_softmax_kernel(
        const unsigned short* __restrict__ Phi, const unsigned short* __restrict__ Plo,
        const unsigned short* __restrict__ Thi, const unsigned short* __restrict__ Tlo,
        const float* __restrict__ p_sq, const float* __restrict__ t_sq,
        float2* __restrict__ partials) {
    __shared__ unsigned short ldsHi[16 * LDSB_STRIDE];
    __shared__ unsigned short ldsLo[16 * LDSB_STRIDE];

    const int wave  = threadIdx.x >> 5;
    const int lane  = threadIdx.x & 31;
    const int half  = lane >> 4;
    const int ln    = lane & 15;
    const int rbase = blockIdx.x * 128 + wave * 16;
    const int cbase = blockIdx.y * COLS_PER_CHUNK;

    // A strip: 16 rows x 128 K as 4 k-blocks of 16x32 bf16 fragments (hi & lo).
    // 16-bit A layout: lanes 0-15 hold K={0..7, 16..23}, lanes 16-31 hold K={8..15, 24..31}.
    FragBF a_hi[4], a_lo[4];
    const size_t abase = (size_t)(rbase + ln) * DDIM;
#pragma unroll
    for (int kb = 0; kb < 4; ++kb) {
        const size_t o = abase + kb * 32 + half * 8;
        a_hi[kb].f[0] = *(const float4*)(Phi + o);
        a_hi[kb].f[1] = *(const float4*)(Phi + o + 16);
        a_lo[kb].f[0] = *(const float4*)(Plo + o);
        a_lo[kb].f[1] = *(const float4*)(Plo + o + 16);
    }

    // per-lane softmax state; C layout: VGPR v <-> row rbase + v + half*8, col = ln
    float ps[8], m[8], s[8];
#pragma unroll
    for (int v = 0; v < 8; ++v) {
        ps[v] = p_sq[rbase + v + half * 8];
        m[v]  = NEG_BIG;
        s[v]  = 0.0f;
    }

    // staging addresses for this thread
    const int col = threadIdx.x >> 4;         // 0..15
    const int k0  = (threadIdx.x & 15) * 8;   // 0..120
    // prefetch tile 0 into registers
    float4 pf_h = *(const float4*)(Thi + (size_t)(cbase + col) * DDIM + k0);
    float4 pf_l = *(const float4*)(Tlo + (size_t)(cbase + col) * DDIM + k0);

    for (int t = 0; t < NTILES; ++t) {
        const int jbase = cbase + t * 16;
        __syncthreads();  // previous tile's LDS reads done before overwrite
        *(float4*)&ldsHi[col * LDSB_STRIDE + k0] = pf_h;
        *(float4*)&ldsLo[col * LDSB_STRIDE + k0] = pf_l;
        __syncthreads();

        if (t + 1 < NTILES) {  // prefetch next tile; latency hidden behind WMMAs
            const size_t g = (size_t)(jbase + 16 + col) * DDIM + k0;
            pf_h = *(const float4*)(Thi + g);
            pf_l = *(const float4*)(Tlo + g);
        }

        // two independent 6-deep WMMA accumulation chains
        v8f acc0 = {}, acc1 = {};
#pragma unroll
        for (int kb = 0; kb < 4; ++kb) {
            // 16-bit B layout: lanes 0-15 hold K={0..7, 16..23}, lanes 16-31 hold K={8..15, 24..31}
            const int bbase = ln * LDSB_STRIDE + kb * 32 + half * 8;
            FragBF bh, bl;
            bh.f[0] = *(const float4*)&ldsHi[bbase];
            bh.f[1] = *(const float4*)&ldsHi[bbase + 16];
            bl.f[0] = *(const float4*)&ldsLo[bbase];
            bl.f[1] = *(const float4*)&ldsLo[bbase + 16];
            if (kb < 2) {
                acc0 = __builtin_amdgcn_wmma_f32_16x16x32_bf16(
                    false, a_hi[kb].v, false, bh.v, (short)0, acc0, false, false);
                acc0 = __builtin_amdgcn_wmma_f32_16x16x32_bf16(
                    false, a_hi[kb].v, false, bl.v, (short)0, acc0, false, false);
                acc0 = __builtin_amdgcn_wmma_f32_16x16x32_bf16(
                    false, a_lo[kb].v, false, bh.v, (short)0, acc0, false, false);
            } else {
                acc1 = __builtin_amdgcn_wmma_f32_16x16x32_bf16(
                    false, a_hi[kb].v, false, bh.v, (short)0, acc1, false, false);
                acc1 = __builtin_amdgcn_wmma_f32_16x16x32_bf16(
                    false, a_hi[kb].v, false, bl.v, (short)0, acc1, false, false);
                acc1 = __builtin_amdgcn_wmma_f32_16x16x32_bf16(
                    false, a_lo[kb].v, false, bh.v, (short)0, acc1, false, false);
            }
        }

        const float tq = t_sq[jbase + ln];
#pragma unroll
        for (int v = 0; v < 8; ++v) {
            const float dot = acc0[v] + acc1[v];
            const float sq  = fmaxf(ps[v] + tq - 2.0f * dot, 0.0f);
            const float l   = -INV_TEMP * __builtin_amdgcn_sqrtf(sq);
            online_update(m[v], s[v], l);
        }
    }

    // merge (m,s) across the 16 lanes of each half (masks < 16 stay in-half)
#pragma unroll
    for (int v = 0; v < 8; ++v) {
#pragma unroll
        for (int mask = 1; mask < 16; mask <<= 1) {
            const float om = __shfl_xor(m[v], mask);
            const float os = __shfl_xor(s[v], mask);
            online_merge(m[v], s[v], om, os);
        }
    }

    if (ln == 0) {  // lanes 0 and 16 cover rows rbase+0..7 and rbase+8..15
#pragma unroll
        for (int v = 0; v < 8; ++v) {
            const int row = rbase + v + half * 8;
            partials[(size_t)blockIdx.y * BSZ + row] = make_float2(m[v], s[v]);
        }
    }
}

// ---------------- kernel 4: merge chunk partials, diag, mean ----------------
__global__ __launch_bounds__(256) void finalize_kernel(const float* __restrict__ P,
                                                       const float* __restrict__ T,
                                                       const float2* __restrict__ partials,
                                                       float* __restrict__ out) {
    const int wave = threadIdx.x >> 5;
    const int lane = threadIdx.x & 31;
    const int row  = blockIdx.x * 8 + wave;

    float m = NEG_BIG, s = 0.0f;
    if (lane < NCHUNK) {
        const float2 pr = partials[(size_t)lane * BSZ + row];
        m = pr.x;
        s = pr.y;
    }
#pragma unroll
    for (int mask = 1; mask < 32; mask <<= 1) {
        const float om = __shfl_xor(m, mask);
        const float os = __shfl_xor(s, mask);
        online_merge(m, s, om, os);
    }

    const float4 p = ((const float4*)(P + (size_t)row * DDIM))[lane];
    const float4 t = ((const float4*)(T + (size_t)row * DDIM))[lane];
    const float dx = p.x - t.x, dy = p.y - t.y, dz = p.z - t.z, dw = p.w - t.w;
    float dd = dx * dx + dy * dy + dz * dz + dw * dw;
#pragma unroll
    for (int mask = 16; mask >= 1; mask >>= 1) dd += __shfl_xor(dd, mask);

    if (lane == 0) {
        const float lse  = m + __logf(s);
        const float ldia = -INV_TEMP * __builtin_amdgcn_sqrtf(fmaxf(dd, 0.0f));
        atomicAdd(out, (lse - ldia) * (1.0f / (float)BSZ));
    }
}

extern "C" void kernel_launch(void* const* d_in, const int* in_sizes, int n_in,
                              void* d_out, int out_size, void* d_ws, size_t ws_size,
                              hipStream_t stream) {
    const float* P = (const float*)d_in[0];
    const float* T = (const float*)d_in[1];
    float* out = (float*)d_out;

    char* ws = (char*)d_ws;
    float*          p_sq     = (float*)(ws);                       // 32 KB
    float*          t_sq     = (float*)(ws + 32768);               // 32 KB
    float2*         partials = (float2*)(ws + 65536);              // 1 MB
    unsigned short* Phi      = (unsigned short*)(ws + 1114112);    // 2 MB each
    unsigned short* Plo      = Phi + (size_t)BSZ * DDIM;
    unsigned short* Thi      = Plo + (size_t)BSZ * DDIM;
    unsigned short* Tlo      = Thi + (size_t)BSZ * DDIM;

    const int nelem4 = (BSZ * DDIM) / 4;
    zero_out_kernel<<<1, 32, 0, stream>>>(out);
    split_bf16_kernel<<<nelem4 / 256, 256, 0, stream>>>(P, Phi, Plo);
    split_bf16_kernel<<<nelem4 / 256, 256, 0, stream>>>(T, Thi, Tlo);
    row_norms_kernel<<<BSZ / 8, 256, 0, stream>>>(P, p_sq);
    row_norms_kernel<<<BSZ / 8, 256, 0, stream>>>(T, t_sq);

    dim3 grid(BSZ / 128, NCHUNK);
    gemm_softmax_kernel<<<grid, 256, 0, stream>>>(Phi, Plo, Thi, Tlo, p_sq, t_sq, partials);

    finalize_kernel<<<BSZ / 8, 256, 0, stream>>>(P, T, partials, out);
}
